// ToyBEVFormer_53747220742151
// MI455X (gfx1250) — compile-verified, matrix operand order checked by reading, not run on hardware
//
#include <hip/hip_runtime.h>
#include <hip/hip_bf16.h>
#include <math.h>

typedef _Float16 v16h __attribute__((ext_vector_type(16)));
typedef _Float16 v8h  __attribute__((ext_vector_type(8)));
typedef float    v8f  __attribute__((ext_vector_type(8)));

#define NQ      10000
#define CDIM    256
#define NCAMS   6
#define HIMG    60
#define WIMG    100
#define NHEADS  8
#define HDCH    32      // channels per head
#define SRAD    0.12f

// ---------------------------------------------------------------------------
// CDNA5 async global->LDS copy (ASYNCcnt path), inline asm: portable across
// both toolchains (bypasses the arity-divergent clang builtin).
// ---------------------------------------------------------------------------
__device__ __forceinline__ void async_copy_b128(unsigned lds_byte_addr,
                                                const void* gaddr) {
    asm volatile("global_load_async_to_lds_b128 %0, %1, off"
                 :: "v"(lds_byte_addr), "v"(gaddr)
                 : "memory");
}
__device__ __forceinline__ void wait_asynccnt0() {
    asm volatile("s_wait_asynccnt 0" ::: "memory");
}
__device__ __forceinline__ unsigned lds_addr_of(const void* p) {
    // LDS aperture flat address: low 32 bits == wave-relative LDS offset
    return (unsigned)(unsigned long long)p;
}

// ---------------------------------------------------------------------------
// f32 -> f16 conversion (grid-stride)
// ---------------------------------------------------------------------------
__global__ void cvt_f32_to_f16(const float* __restrict__ in,
                               _Float16* __restrict__ out, int n) {
    int i = blockIdx.x * blockDim.x + threadIdx.x;
    int stride = gridDim.x * blockDim.x;
    for (; i < n; i += stride) out[i] = (_Float16)in[i];
}

// ---------------------------------------------------------------------------
// WMMA GEMM:  C[M x ncols] = A[M x 256] * B[256 x ncols] + bias
// f16 inputs, f32 accumulate.  8 waves/block; block tile 128 rows x 64 cols;
// each wave owns 16 rows x 64 cols (4 accumulators sharing one A fragment).
// A panel async-staged to LDS (no transpose needed); B tile transpose-staged.
// mode 0: tanh(x)*SRAD epilogue, else identity.  M = NQ = 625 exact tiles.
// ---------------------------------------------------------------------------
__global__ void wmma_gemm_f16(const _Float16* __restrict__ A,
                              const _Float16* __restrict__ B,
                              const float* __restrict__ bias,
                              float* __restrict__ Cout,
                              int ncols, int mode) {
    __shared__ __align__(16) _Float16 aS[128 * 32];  // [r][k]  8 KB
    __shared__ __align__(16) _Float16 bS[64 * 32];   // [n][k]  4 KB (transposed)

    const int tid  = threadIdx.x;
    const int lane = tid & 31;
    const int wave = tid >> 5;
    const int half = lane >> 4;               // K-half selector (ISA layout)
    const int mn   = lane & 15;               // row (A) / col (B,C) index
    const int tile = blockIdx.x * 8 + wave;   // 16-row tile
    const int row0 = blockIdx.x * 128;
    const int col0 = blockIdx.y * 64;
    const bool valid = (tile * 16 < NQ);      // wave-uniform -> EXEC stays full

    const unsigned aBase = lds_addr_of(&aS[0]);

    v8f acc0 = {}, acc1 = {}, acc2 = {}, acc3 = {};

    for (int k0 = 0; k0 < CDIM; k0 += 32) {
        // ---- async-stage A panel chunk: 128 rows x 32 k (512 x 16B chunks)
        #pragma unroll
        for (int i = 0; i < 2; ++i) {
            int ci = tid + i * 256;           // chunk id 0..511
            int r  = ci >> 2;                 // row within block tile
            int s  = ci & 3;                  // 8-element K segment
            if (row0 + r < NQ) {
                async_copy_b128(aBase + (unsigned)ci * 16u,
                                A + (size_t)(row0 + r) * CDIM + k0 + s * 8);
            }
        }
        // ---- transpose-stage B tile: 32 k x 64 cols -> bS[n][k]
        #pragma unroll
        for (int i = 0; i < 8; ++i) {
            int idx = tid + i * 256;          // 0..2047
            int nc  = idx & 63;
            int kk  = idx >> 6;
            bS[nc * 32 + kk] = B[(size_t)(k0 + kk) * ncols + col0 + nc];
            if (k0 + 32 < CDIM)               // prefetch next B tile (L2 hint)
                __builtin_prefetch(&B[(size_t)(k0 + 32 + kk) * ncols + col0 + nc], 0, 1);
        }
        wait_asynccnt0();
        __syncthreads();

        if (valid) {
            // A fragment: lane holds row mn of its wave tile
            const _Float16* ar = &aS[(wave * 16 + mn) * 32];
            v8h alo = *(const v8h*)(ar + 8 * half);
            v8h ahi = *(const v8h*)(ar + 16 + 8 * half);
            v16h av = __builtin_shufflevector(alo, ahi,
                        0,1,2,3,4,5,6,7,8,9,10,11,12,13,14,15);
            #pragma unroll
            for (int j = 0; j < 4; ++j) {
                const _Float16* br = &bS[(j * 16 + mn) * 32];
                v8h blo = *(const v8h*)(br + 8 * half);
                v8h bhi = *(const v8h*)(br + 16 + 8 * half);
                v16h bv = __builtin_shufflevector(blo, bhi,
                            0,1,2,3,4,5,6,7,8,9,10,11,12,13,14,15);
                v8f* ac = (j == 0) ? &acc0 : (j == 1) ? &acc1
                                           : (j == 2) ? &acc2 : &acc3;
                *ac = __builtin_amdgcn_wmma_f32_16x16x32_f16(
                          false, av, false, bv, (short)0, *ac, false, false);
            }
        }
        __syncthreads();
    }

    if (valid) {
        #pragma unroll
        for (int j = 0; j < 4; ++j) {
            const v8f* ac = (j == 0) ? &acc0 : (j == 1) ? &acc1
                                             : (j == 2) ? &acc2 : &acc3;
            int col = col0 + j * 16 + mn;
            float bval = bias[col];
            #pragma unroll
            for (int v = 0; v < 8; ++v) {
                int row = tile * 16 + 8 * half + v;  // lanes>=16 -> M = 8+v
                float x = (*ac)[v] + bval;
                if (mode == 0) x = tanhf(x) * SRAD;
                Cout[(size_t)row * ncols + col] = x;
            }
        }
    }
}

// ---------------------------------------------------------------------------
// Per-query sampling + masked softmax + camera fusion.  One block per query.
// Phase 1 (tid<128, one per (head,point)): masked logits, bilinear coords.
// Phase 2 (all 256, one per (head,channel)): weighted bilinear gather.
// Image feats (37 MB) are L2-resident on MI455X (192 MB L2).
// ---------------------------------------------------------------------------
__global__ void sample_fuse(const float* __restrict__ offsets,   // [N,256]
                            const float* __restrict__ logits,    // [N,128]
                            const float* __restrict__ img,       // [6,256,60,100]
                            const float* __restrict__ refp,      // [6,N,4,2]
                            const int*   __restrict__ mask,      // [6,N,4]
                            _Float16* __restrict__ fused16)      // [N,256]
{
    __shared__ float slog[128];
    __shared__ float sw[128];
    __shared__ int   sx0[128];
    __shared__ int   sy0[128];
    __shared__ float swx[128];
    __shared__ float swy[128];
    __shared__ int   svis[4];

    const int n = blockIdx.x;
    const int tid = threadIdx.x;
    float accum = 0.f;
    int count = 0;

    for (int cam = 0; cam < NCAMS; ++cam) {
        if (tid < 4) svis[tid] = mask[((size_t)cam * NQ + n) * 4 + tid];
        __syncthreads();
        const bool hv = (svis[0] | svis[1] | svis[2] | svis[3]) != 0;

        if (hv && tid < 128) {
            int hh = tid >> 4, p = tid & 15;
            int pillar = p >> 2, pt = p & 3;
            float l = logits[(size_t)n * 128 + tid];
            slog[tid] = svis[pillar] ? l : -1e9f;
            const float* rp = refp + (((size_t)cam * NQ + n) * 4 + pillar) * 2;
            int oidx = ((hh * 4 + pillar) * 4 + pt) * 2;
            float x = (rp[0] + offsets[(size_t)n * 256 + oidx + 0]) * WIMG - 0.5f;
            float y = (rp[1] + offsets[(size_t)n * 256 + oidx + 1]) * HIMG - 0.5f;
            float x0f = floorf(x), y0f = floorf(y);
            sx0[tid] = (int)x0f;  sy0[tid] = (int)y0f;
            swx[tid] = x - x0f;   swy[tid] = y - y0f;
        }
        __syncthreads();
        if (hv && tid < 128) {
            int p = tid & 15;
            int pillar = p >> 2;
            const float* row = &slog[tid & ~15];
            float mx = -1e9f;
            #pragma unroll
            for (int j = 0; j < 16; ++j) mx = fmaxf(mx, row[j]);
            float s = 0.f;
            #pragma unroll
            for (int j = 0; j < 16; ++j) s += __expf(row[j] - mx);
            sw[tid] = svis[pillar] ? __expf(slog[tid] - mx) / fmaxf(s, 1e-6f)
                                   : 0.f;
        }
        __syncthreads();
        if (hv) {
            int hh = tid >> 5, c = tid & 31;
            const float* feat =
                img + ((size_t)cam * CDIM + (hh * HDCH + c)) * (HIMG * WIMG);
            float camacc = 0.f;
            int base = hh * 16;
            #pragma unroll 4
            for (int p = 0; p < 16; ++p) {
                float w = sw[base + p];
                if (w == 0.f) continue;
                int   x0 = sx0[base + p], y0 = sy0[base + p];
                float wx = swx[base + p], wy = swy[base + p];
                bool xv0 = (x0 >= 0) && (x0 < WIMG);
                bool xv1 = (x0 + 1 >= 0) && (x0 + 1 < WIMG);
                float s = 0.f;
                if (y0 >= 0 && y0 < HIMG) {
                    const float* r = feat + (size_t)y0 * WIMG;
                    if (xv0) s += r[x0]     * (1.f - wx) * (1.f - wy);
                    if (xv1) s += r[x0 + 1] * wx * (1.f - wy);
                }
                if (y0 + 1 >= 0 && y0 + 1 < HIMG) {
                    const float* r = feat + (size_t)(y0 + 1) * WIMG;
                    if (xv0) s += r[x0]     * (1.f - wx) * wy;
                    if (xv1) s += r[x0 + 1] * wx * wy;
                }
                camacc += w * s;
            }
            accum += camacc;
            count += 1;
        }
        __syncthreads();
    }
    float denom = (float)(count < 1 ? 1 : count);
    int hh = tid >> 5, c = tid & 31;
    fused16[(size_t)n * 256 + hh * HDCH + c] = (_Float16)(accum / denom);
}

// ---------------------------------------------------------------------------
extern "C" void kernel_launch(void* const* d_in, const int* in_sizes, int n_in,
                              void* d_out, int out_size, void* d_ws, size_t ws_size,
                              hipStream_t stream) {
    const float* query = (const float*)d_in[0];
    const float* img   = (const float*)d_in[1];
    const float* refp  = (const float*)d_in[2];
    const int*   bmask = (const int*)d_in[3];
    const float* offW  = (const float*)d_in[4];
    const float* offb  = (const float*)d_in[5];
    const float* wW    = (const float*)d_in[6];
    const float* wb    = (const float*)d_in[7];
    const float* outW  = (const float*)d_in[8];
    const float* outb  = (const float*)d_in[9];
    float* out = (float*)d_out;

    char* ws = (char*)d_ws;
    _Float16* q16     = (_Float16*)ws;  ws += (size_t)NQ * 256 * 2;
    _Float16* offW16  = (_Float16*)ws;  ws += (size_t)256 * 256 * 2;
    _Float16* wW16    = (_Float16*)ws;  ws += (size_t)256 * 128 * 2;
    _Float16* outW16  = (_Float16*)ws;  ws += (size_t)256 * 256 * 2;
    _Float16* fused16 = (_Float16*)ws;  ws += (size_t)NQ * 256 * 2;
    float*    offsets = (float*)ws;     ws += (size_t)NQ * 256 * 4;
    float*    wlogits = (float*)ws;     ws += (size_t)NQ * 128 * 4;

    // 1) f16 conversions
    cvt_f32_to_f16<<<2048, 256, 0, stream>>>(query, q16, NQ * 256);
    cvt_f32_to_f16<<<256, 256, 0, stream>>>(offW, offW16, 256 * 256);
    cvt_f32_to_f16<<<128, 256, 0, stream>>>(wW, wW16, 256 * 128);
    cvt_f32_to_f16<<<256, 256, 0, stream>>>(outW, outW16, 256 * 256);

    const int gx = (NQ / 16 + 7) / 8;             // 79 row blocks (128 rows each)

    // 2) projections (WMMA):  offsets = tanh(q@offW+b)*0.12 ; logits = q@wW+b
    wmma_gemm_f16<<<dim3(gx, 256 / 64), 256, 0, stream>>>(q16, offW16, offb,
                                                          offsets, 256, 0);
    wmma_gemm_f16<<<dim3(gx, 128 / 64), 256, 0, stream>>>(q16, wW16, wb,
                                                          wlogits, 128, 1);

    // 3) masked softmax + bilinear sampling + camera fusion
    sample_fuse<<<NQ, 256, 0, stream>>>(offsets, wlogits, img, refp, bmask,
                                        fused16);

    // 4) output projection (WMMA) -> d_out (f32)
    wmma_gemm_f16<<<dim3(gx, 256 / 64), 256, 0, stream>>>(fused16, outW16, outb,
                                                          out, 256, 1);
}
